// EquiGNN_55628416418478
// MI455X (gfx1250) — matrix operand order repaired and checked by reading, not compiled, not run on hardware
//
#include <hip/hip_runtime.h>
#include <hip/hip_bf16.h>
#include <initializer_list>
#include <stdint.h>

// ---------------------------------------------------------------------------
// EGNN forward for MI455X (gfx1250, wave32, WMMA).
// - all GEMMs via v_wmma_f32_16x16x32_bf16 (bf16 in, f32 acc)
// - weights stored TRANSPOSED (N x Kpad) so each lane's B fragment is two contiguous 16B loads
// - RBF featurization + concats fused into the A-operand fetch
// - NT=4 column tiles per wave: per K-step, all B fragments are preloaded
//   (one load clause, one wait), then 4 WMMAs issue back-to-back
// - LDS-staged weights via global_load_async_to_lds_b128 when available
// ---------------------------------------------------------------------------

typedef __bf16 bf16;
typedef __attribute__((ext_vector_type(16))) __bf16 v16bf;
typedef __attribute__((ext_vector_type(8)))  __bf16 v8bf;
typedef __attribute__((ext_vector_type(8)))  float  v8f;
typedef int v4i_gcc __attribute__((__vector_size__(16)));  // int4 for async-LDS builtin

static_assert(sizeof(bf16) == 2, "bf16 must be 2 bytes");

__device__ __forceinline__ float fast_rcp(float x) {
#if __has_builtin(__builtin_amdgcn_rcpf)
  return __builtin_amdgcn_rcpf(x);
#else
  return 1.f / x;
#endif
}

// Segment kinds for the A-operand of the generic GEMM.
//  0: matrix, row = r                      (plain)
//  1: matrix, row = r>>5                   (h_i broadcast over pair rows)
//  2: matrix, row = (r>>10)<<5 | (r&31)    (h_j broadcast over pair rows)
//  3: RBF(dist)  from f32 radial[r], pair rows  (32 centers + time + pad to 64)
//  4: RBF(angle) from f32 angle[r],  tri rows   (32 centers + time + pad to 64)
struct Seg {
  const void* p;
  int kind;
  int kcols;      // multiple of 32
  int ld;         // leading dim (bf16 elements) for matrix kinds
  int maskdiag;   // zero the segment on diagonal pair rows (i==j)
};

struct GemmCfg {
  Seg seg[3];
  int nseg;
  int rows;       // multiple of 16
  int Nout;       // multiple of 16 (NT=4 requires multiple of 64)
  int act;        // 0 none, 1 relu, 2 swish
  int mask_pair;  // zero output rows where i==j (pair rows)
  int use_lds;    // stage Wt in LDS
  const bf16* Wt; // Nout x Kpad, row-major bf16 (i.e. transposed weights)
  const float* bias;  // length Nout or null
  float* outf;    // f32 out or null
  bf16* outb;     // bf16 out or null
  const float* tvec;  // t per batch (16)
};

// A fragment: 16x32 bf16 tile, ISA layout:
// lanes 0-15: M=lane, halves 0..7 -> K=0..7,  halves 8..15 -> K=16..23
// lanes16-31: M=lane-16, halves 0..7 -> K=8..15, halves 8..15 -> K=24..31
__device__ __forceinline__ v16bf load_A(const GemmCfg& c, int rowBase, int lane,
                                        int kt) {
  const int r = rowBase + (lane & 15);
  const int kgrp = (lane >> 4) << 3;  // 0 or 8
  const int k0 = kt << 5;
  int s = 0, segbase = 0;
  while (s + 1 < c.nseg && k0 >= segbase + c.seg[s].kcols) {
    segbase += c.seg[s].kcols;
    ++s;
  }
  const Seg sg = c.seg[s];
  const int kl = k0 - segbase;
  v16bf a;
  if (sg.kind <= 2) {
    int row = r;
    if (sg.kind == 1)      row = r >> 5;
    else if (sg.kind == 2) row = ((r >> 10) << 5) | (r & 31);
    const bf16* p = (const bf16*)sg.p + (size_t)row * sg.ld + kl + kgrp;
    if (sg.maskdiag && (((r >> 5) & 31) == (r & 31))) {
#pragma unroll
      for (int h = 0; h < 16; ++h) a[h] = (bf16)0.f;
    } else {
      __builtin_prefetch(p + 32, 0, 0);  // -> global_prefetch_b8
      v8bf lo = *(const v8bf*)p;         // K = kl+kgrp .. +7
      v8bf hi = *(const v8bf*)(p + 16);  // K = kl+16+kgrp .. +7
#pragma unroll
      for (int h = 0; h < 8; ++h) { a[h] = lo[h]; a[h + 8] = hi[h]; }
    }
  } else {
    float x, tm, coef, step;
    if (sg.kind == 3) {  // dist RBF on pair rows
      int b = r >> 10, i = (r >> 5) & 31, j = r & 31;
      x = ((const float*)sg.p)[r];
      tm = (i != j) ? c.tvec[b] : 0.f;
      coef = -0.5f; step = 30.f / 31.f;
    } else {             // angle RBF on tri rows
      int b = r >> 15, i = (r >> 10) & 31, j = (r >> 5) & 31, kk = r & 31;
      x = ((const float*)sg.p)[r];
      tm = ((i != kk) && (j != kk)) ? c.tvec[b] : 0.f;
      coef = -10.f; step = 0.1f;
    }
#pragma unroll
    for (int h = 0; h < 16; ++h) {
      int kidx = kl + ((h < 8) ? (kgrp + h) : (16 + kgrp + (h - 8)));
      float v = 0.f;
      if (kidx < 32) {
        float d = x - step * (float)kidx;
        v = __expf(coef * d * d);
      } else if (kidx == 32) {
        v = tm;
      }
      a[h] = (bf16)v;
    }
  }
  return a;
}

// B fragment from TRANSPOSED weights Wt[N][Kld]: lane -> col = base+lane%16,
// same K layout as A: lanes 0-15 K=0..7,16..23; lanes 16-31 K=8..15,24..31,
// i.e. two 16B loads.
template <int KT>
__device__ __forceinline__ v16bf load_Bt(const bf16* Wt, int colBase, int lane,
                                         int kt) {
  const int Kld = KT * 32;
  const int col = colBase + (lane & 15);
  const int kb = (kt << 5) + ((lane >> 4) << 3);
  const bf16* p = Wt + (size_t)col * Kld + kb;
  v8bf lo = *(const v8bf*)p;
  v8bf hi = *(const v8bf*)(p + 16);
  v16bf b;
#pragma unroll
  for (int h = 0; h < 8; ++h) { b[h] = lo[h]; b[h + 8] = hi[h]; }
  return b;
}

template <int KT, int NT>
__global__ __launch_bounds__(256) void k_wmma_gemm(GemmCfg c) {
  extern __shared__ bf16 lW[];
  // local copies -> keep in SGPRs, avoid kernarg reloads in inner loops
  float* const of = c.outf;
  bf16* const ob = c.outb;
  const float* const bias = c.bias;
  const int cact = c.act;
  const int cmask = c.mask_pair;
  const int NoutL = c.Nout;
  const int useLds = c.use_lds;

  if (useLds) {
    const int nv = (KT * 32 * NoutL) >> 3;  // 16B chunks
    const v4i_gcc* src = (const v4i_gcc*)c.Wt;
    v4i_gcc* dst = (v4i_gcc*)lW;
#if __has_builtin(__builtin_amdgcn_global_load_async_to_lds_b128) && \
    __has_builtin(__builtin_amdgcn_s_wait_asynccnt)
    for (int idx = threadIdx.x; idx < nv; idx += blockDim.x) {
      __builtin_amdgcn_global_load_async_to_lds_b128(
          (__attribute__((address_space(1))) v4i_gcc*)(uintptr_t)(src + idx),
          (__attribute__((address_space(3))) v4i_gcc*)(uint32_t)(uintptr_t)(dst + idx),
          0, 0);
    }
    __builtin_amdgcn_s_wait_asynccnt(0);
#else
    for (int idx = threadIdx.x; idx < nv; idx += blockDim.x) dst[idx] = src[idx];
#endif
    __syncthreads();
  }
  const int lane = threadIdx.x & 31;
  const int wid = blockIdx.x * (blockDim.x >> 5) + (threadIdx.x >> 5);
  const int colGroups = NoutL / (16 * NT);
  if (wid >= (c.rows >> 4) * colGroups) return;
  const int rowBase = (wid / colGroups) << 4;
  const int colBase0 = (wid % colGroups) * (16 * NT);

  v8f acc[NT];
#pragma unroll
  for (int n = 0; n < NT; ++n)
    acc[n] = (v8f){0.f, 0.f, 0.f, 0.f, 0.f, 0.f, 0.f, 0.f};

  // use_lds branch hoisted OUT of the K-loop; per kt: load A, preload all NT
  // B fragments (single clause + wait), then NT back-to-back WMMAs (disjoint
  // accumulators, shared A -> no D->A/B RAW hazard).
  if (useLds) {
#pragma unroll
    for (int kt = 0; kt < KT; ++kt) {
      v16bf a = load_A(c, rowBase, lane, kt);
      v16bf bf[NT];
#pragma unroll
      for (int n = 0; n < NT; ++n)
        bf[n] = load_Bt<KT>(lW, colBase0 + n * 16, lane, kt);
#pragma unroll
      for (int n = 0; n < NT; ++n)
        acc[n] = __builtin_amdgcn_wmma_f32_16x16x32_bf16(false, a, false, bf[n],
                                                         (short)0, acc[n],
                                                         false, false);
    }
  } else {
#pragma unroll
    for (int kt = 0; kt < KT; ++kt) {
      v16bf a = load_A(c, rowBase, lane, kt);
      v16bf bf[NT];
#pragma unroll
      for (int n = 0; n < NT; ++n)
        bf[n] = load_Bt<KT>(c.Wt, colBase0 + n * 16, lane, kt);
#pragma unroll
      for (int n = 0; n < NT; ++n)
        acc[n] = __builtin_amdgcn_wmma_f32_16x16x32_bf16(false, a, false, bf[n],
                                                         (short)0, acc[n],
                                                         false, false);
    }
  }

  // ---- epilogue: all mode branches hoisted & uniform ----
  if (bias) {
#pragma unroll
    for (int n = 0; n < NT; ++n) {
      const float bv = bias[colBase0 + n * 16 + (lane & 15)];
#pragma unroll
      for (int rr = 0; rr < 8; ++rr) acc[n][rr] += bv;
    }
  }
  if (cact == 1) {
#pragma unroll
    for (int n = 0; n < NT; ++n)
#pragma unroll
      for (int rr = 0; rr < 8; ++rr) acc[n][rr] = fmaxf(acc[n][rr], 0.f);
  } else if (cact == 2) {
#pragma unroll
    for (int n = 0; n < NT; ++n)
#pragma unroll
      for (int rr = 0; rr < 8; ++rr) {
        float v = acc[n][rr];
        acc[n][rr] = v * fast_rcp(1.f + __expf(-v));
      }
  }
  if (cmask) {
#pragma unroll
    for (int rr = 0; rr < 8; ++rr) {
      int row = rowBase + rr + ((lane >> 4) << 3);
      float f = (((row >> 5) & 31) == (row & 31)) ? 0.f : 1.f;
#pragma unroll
      for (int n = 0; n < NT; ++n) acc[n][rr] *= f;
    }
  }
  const int col0 = colBase0 + (lane & 15);
  const int rb = rowBase + ((lane >> 4) << 3);
  if (of) {
#pragma unroll
    for (int n = 0; n < NT; ++n)
#pragma unroll
      for (int rr = 0; rr < 8; ++rr)
        of[(size_t)(rb + rr) * NoutL + col0 + n * 16] = acc[n][rr];
  }
  if (ob) {
#pragma unroll
    for (int n = 0; n < NT; ++n)
#pragma unroll
      for (int rr = 0; rr < 8; ++rr)
        ob[(size_t)(rb + rr) * NoutL + col0 + n * 16] = (bf16)acc[n][rr];
  }
}

// ---------------- small support kernels ----------------

// Convert f32 weight (srows x scols) -> bf16 TRANSPOSED, zero-padded:
// dst[n * drows + k] = src[k * scols + n], dst is (dcols x drows).
__global__ void k_cvtT(const float* s, bf16* d, int srows, int scols, int drows,
                       int dcols) {
  int id = blockIdx.x * blockDim.x + threadIdx.x;
  if (id >= drows * dcols) return;
  int n = id / drows, k = id - n * drows;
  float v = (k < srows && n < scols) ? s[k * scols + n] : 0.f;
  d[id] = (bf16)v;
}

__global__ void k_embed(const float* z, const float* t, float* pos0, float* pos,
                        bf16* X) {
  int row = blockIdx.x * blockDim.x + threadIdx.x;
  if (row >= 512) return;
  int b = row >> 5;
#pragma unroll
  for (int d = 0; d < 3; ++d) {
    float v = z[row * 10 + d];
    pos0[row * 3 + d] = v;
    pos[row * 3 + d] = v;
  }
#pragma unroll
  for (int k = 0; k < 32; ++k) {
    float v = 0.f;
    if (k < 7)       v = z[row * 10 + 3 + k];
    else if (k == 7) v = t[b];
    X[row * 32 + k] = (bf16)v;
  }
}

__global__ void k_radial(const float* pos, float* radial, float* cd, float* vn) {
  int r = blockIdx.x * blockDim.x + threadIdx.x;
  if (r >= 16384) return;
  int pi = (r >> 5) * 3;                              // b*32+i
  int pj = (((r >> 10) << 5) | (r & 31)) * 3;         // b*32+j
  float dx = pos[pi] - pos[pj], dy = pos[pi + 1] - pos[pj + 1],
        dz = pos[pi + 2] - pos[pj + 2];
  float rad = dx * dx + dy * dy + dz * dz;
  radial[r] = rad;
  float inv = fast_rcp(sqrtf(rad + 1e-8f) + 1.f);
  cd[r * 3] = dx * inv; cd[r * 3 + 1] = dy * inv; cd[r * 3 + 2] = dz * inv;
  float ivn = fast_rcp(sqrtf(rad) + 1e-8f);           // vn uses pos_j - pos_i
  vn[r * 3] = -dx * ivn; vn[r * 3 + 1] = -dy * ivn; vn[r * 3 + 2] = -dz * ivn;
}

__global__ void k_angle(const float* vn, float* ang) {
  int r = blockIdx.x * blockDim.x + threadIdx.x;
  if (r >= 524288) return;
  int bi = r >> 10;                 // b*32+i
  int j = (r >> 5) & 31, k = r & 31;
  int p1 = ((bi << 5) + j) * 3, p2 = ((bi << 5) + k) * 3;
  float c = vn[p1] * vn[p2] + vn[p1 + 1] * vn[p2 + 1] + vn[p1 + 2] * vn[p2 + 2];
  c = fminf(fmaxf(c, -1.f + 1e-6f), 1.f - 1e-6f);
  ang[r] = acosf(c);
}

__global__ void k_tsum(const bf16* tri, bf16* ts) {
  int id = blockIdx.x * blockDim.x + threadIdx.x;
  if (id >= 16384 * 64) return;
  int p = id >> 6, e = id & 63;
  const bf16* base = tri + (size_t)p * 32 * 64 + e;
  float s = 0.f;
#pragma unroll
  for (int k = 0; k < 32; ++k) s += (float)base[k * 64];
  ts[id] = (bf16)s;
}

__global__ void k_pairnew(const float* m, const float* tl, bf16* pair) {
  int id = blockIdx.x * blockDim.x + threadIdx.x;
  if (id >= 16384 * 128) return;
  int r = id >> 7;
  float y = m[id] + tl[id];
  if (((r >> 5) & 31) == (r & 31)) y = 0.f;
  pair[id] = (bf16)y;
}

__global__ void k_agg(const bf16* pair, bf16* agg) {
  int id = blockIdx.x * blockDim.x + threadIdx.x;
  if (id >= 512 * 128) return;
  int bi = id >> 7, e = id & 127;
  float s = 0.f;
#pragma unroll
  for (int j = 0; j < 32; ++j)
    s += (float)pair[(((size_t)bi << 5) + j) * 128 + e];
  agg[id] = (bf16)s;
}

__global__ __launch_bounds__(32) void k_coord(const float* m, const float* cd,
                                              const float* cw, const float* cb,
                                              float* pos) {
  int bi = blockIdx.x;          // b*32+i
  int j = threadIdx.x;
  int i = bi & 31;
  size_t row = (size_t)bi * 32 + j;
  float w = cb[0];
  const float* mp = m + row * 128;
  for (int e = 0; e < 128; ++e) w += mp[e] * cw[e];
  float mask = (i != j) ? 1.f : 0.f;
  float sx = cd[row * 3] * w * mask;
  float sy = cd[row * 3 + 1] * w * mask;
  float sz = cd[row * 3 + 2] * w * mask;
#pragma unroll
  for (int off = 16; off > 0; off >>= 1) {
    sx += __shfl_down(sx, off);
    sy += __shfl_down(sy, off);
    sz += __shfl_down(sz, off);
  }
  if (j == 0) {
    pos[bi * 3]     += sx * (1.f / 31.f);
    pos[bi * 3 + 1] += sy * (1.f / 31.f);
    pos[bi * 3 + 2] += sz * (1.f / 31.f);
  }
}

__global__ void k_hupd(float* hf, const float* hd, bf16* hb) {
  int id = blockIdx.x * blockDim.x + threadIdx.x;
  if (id >= 512 * 128) return;
  float v = hf[id] + hd[id];
  hf[id] = v;
  hb[id] = (bf16)v;
}

__global__ __launch_bounds__(32) void k_velmean(const float* pos,
                                                const float* pos0,
                                                float* mean) {
  int b = blockIdx.x, i = threadIdx.x;
  int r = (b << 5) + i;
  float vx = pos[r * 3] - pos0[r * 3];
  float vy = pos[r * 3 + 1] - pos0[r * 3 + 1];
  float vz = pos[r * 3 + 2] - pos0[r * 3 + 2];
#pragma unroll
  for (int off = 16; off > 0; off >>= 1) {
    vx += __shfl_down(vx, off);
    vy += __shfl_down(vy, off);
    vz += __shfl_down(vz, off);
  }
  if (i == 0) {
    mean[b * 3]     = vx * (1.f / 32.f);
    mean[b * 3 + 1] = vy * (1.f / 32.f);
    mean[b * 3 + 2] = vz * (1.f / 32.f);
  }
}

__global__ void k_final(const float* pos, const float* pos0, const float* mean,
                        const float* xh, float* out) {
  int r = blockIdx.x * blockDim.x + threadIdx.x;
  if (r >= 512) return;
  int b = r >> 5;
#pragma unroll
  for (int d = 0; d < 3; ++d)
    out[r * 10 + d] = (pos[r * 3 + d] - pos0[r * 3 + d]) - mean[b * 3 + d];
#pragma unroll
  for (int f = 0; f < 7; ++f) out[r * 10 + 3 + f] = xh[r * 16 + f];
}

// ---------------- host orchestration ----------------

extern "C" void kernel_launch(void* const* d_in, const int* in_sizes, int n_in,
                              void* d_out, int out_size, void* d_ws,
                              size_t ws_size, hipStream_t stream) {
  (void)in_sizes; (void)n_in; (void)out_size; (void)ws_size;
  // Input layout: z_t, t, node_mask, pair_mask, then params leaves in JAX
  // pytree order (sorted dict keys, 'b' before 'w', None (out3.b) skipped).
  const float* z_t  = (const float*)d_in[0];
  const float* tvec = (const float*)d_in[1];
  const float* angle_b = (const float*)d_in[4];
  const float* angle_w = (const float*)d_in[5];
  const float* edge_b  = (const float*)d_in[6];
  const float* edge_w  = (const float*)d_in[7];
  const float* eef_b   = (const float*)d_in[8];
  const float* eef_w   = (const float*)d_in[9];
  const float *co_b[4], *co_w[4], *e1b[4], *e1w[4], *e2b[4], *e2w[4],
              *n1b[4], *n1w[4], *n2b[4], *n2w[4], *trb[4], *trw[4];
  for (int l = 0; l < 4; ++l) {
    int base = 10 + 12 * l;
    co_b[l] = (const float*)d_in[base + 0];
    co_w[l] = (const float*)d_in[base + 1];
    e1b[l]  = (const float*)d_in[base + 2];
    e1w[l]  = (const float*)d_in[base + 3];
    e2b[l]  = (const float*)d_in[base + 4];
    e2w[l]  = (const float*)d_in[base + 5];
    n1b[l]  = (const float*)d_in[base + 6];
    n1w[l]  = (const float*)d_in[base + 7];
    n2b[l]  = (const float*)d_in[base + 8];
    n2w[l]  = (const float*)d_in[base + 9];
    trb[l]  = (const float*)d_in[base + 10];
    trw[l]  = (const float*)d_in[base + 11];
  }
  const float* npf_b = (const float*)d_in[58];
  const float* npf_w = (const float*)d_in[59];
  const float* o1b = (const float*)d_in[60];
  const float* o1w = (const float*)d_in[61];
  const float* o2b = (const float*)d_in[62];
  const float* o2w = (const float*)d_in[63];
  const float* o3w = (const float*)d_in[64];
  const float* tef_b = (const float*)d_in[65];
  const float* tef_w = (const float*)d_in[66];
  const float* xh_b = (const float*)d_in[67];
  const float* xh_w = (const float*)d_in[68];

  char* wp = (char*)d_ws;
  auto alloc = [&](size_t n) {
    void* r = (void*)wp;
    wp += (n + 255) & ~(size_t)255;
    return r;
  };
  // transpose-convert: src (srows x scols) f32 -> dst (dcols x drows) bf16
  auto cvt = [&](const float* s, bf16* d, int sr, int sc, int dr, int dc) {
    int tot = dr * dc;
    k_cvtT<<<(tot + 255) / 256, 256, 0, stream>>>(s, d, sr, sc, dr, dc);
  };

  // bf16 transposed weights (K padded to multiples of 32, out3 cols pad to 16)
  bf16* Wxh  = (bf16*)alloc(32 * 128 * 2);   cvt(xh_w, Wxh, 8, 128, 32, 128);
  bf16* Wde  = (bf16*)alloc(64 * 128 * 2);   cvt(edge_w, Wde, 33, 128, 64, 128);
  bf16* Wae  = (bf16*)alloc(64 * 64 * 2);    cvt(angle_w, Wae, 33, 64, 64, 64);
  bf16* Wnpf = (bf16*)alloc(384 * 128 * 2);  cvt(npf_w, Wnpf, 384, 128, 384, 128);
  bf16* Weef = (bf16*)alloc(256 * 128 * 2);  cvt(eef_w, Weef, 256, 128, 256, 128);
  bf16* Wtef = (bf16*)alloc(128 * 64 * 2);   cvt(tef_w, Wtef, 128, 64, 128, 64);
  bf16* Wo1  = (bf16*)alloc(128 * 256 * 2);  cvt(o1w, Wo1, 128, 256, 128, 256);
  bf16* Wo2  = (bf16*)alloc(256 * 128 * 2);  cvt(o2w, Wo2, 256, 128, 256, 128);
  bf16* Wo3  = (bf16*)alloc(128 * 16 * 2);   cvt(o3w, Wo3, 128, 8, 128, 16);
  bf16 *We1[4], *We2[4], *Wtri[4], *Wn1[4], *Wn2[4];
  for (int l = 0; l < 4; ++l) {
    We1[l]  = (bf16*)alloc(384 * 256 * 2); cvt(e1w[l], We1[l], 384, 256, 384, 256);
    We2[l]  = (bf16*)alloc(256 * 128 * 2); cvt(e2w[l], We2[l], 256, 128, 256, 128);
    Wtri[l] = (bf16*)alloc(64 * 128 * 2);  cvt(trw[l], Wtri[l], 64, 128, 64, 128);
    Wn1[l]  = (bf16*)alloc(256 * 256 * 2); cvt(n1w[l], Wn1[l], 256, 256, 256, 256);
    Wn2[l]  = (bf16*)alloc(256 * 128 * 2); cvt(n2w[l], Wn2[l], 256, 128, 256, 128);
  }

  // activations / scratch
  float* pos0   = (float*)alloc(512 * 3 * 4);
  float* pos    = (float*)alloc(512 * 3 * 4);
  bf16*  Xemb   = (bf16*)alloc(512 * 32 * 2);
  float* h_f    = (float*)alloc(512 * 128 * 4);
  bf16*  h_bm   = (bf16*)alloc(512 * 128 * 2);
  float* radial = (float*)alloc(16384 * 4);
  float* cd     = (float*)alloc(16384 * 3 * 4);
  float* vn     = (float*)alloc(16384 * 3 * 4);
  float* ang    = (float*)alloc((size_t)524288 * 4);
  bf16*  dist_b = (bf16*)alloc((size_t)16384 * 128 * 2);
  bf16*  pair_b = (bf16*)alloc((size_t)16384 * 128 * 2);
  bf16*  pe1_b  = (bf16*)alloc((size_t)16384 * 128 * 2);
  bf16*  e1_act = (bf16*)alloc((size_t)16384 * 256 * 2);
  float* m_f    = (float*)alloc((size_t)16384 * 128 * 4);
  float* tl_f   = (float*)alloc((size_t)16384 * 128 * 4);
  bf16*  ts_b   = (bf16*)alloc((size_t)16384 * 64 * 2);
  bf16*  agg_b  = (bf16*)alloc(512 * 128 * 2);
  bf16*  n1_act = (bf16*)alloc(512 * 256 * 2);
  float* hd_f   = (float*)alloc(512 * 128 * 4);
  bf16*  tri_a  = (bf16*)alloc((size_t)524288 * 64 * 2);
  bf16*  tri_b2 = (bf16*)alloc((size_t)524288 * 64 * 2);
  bf16*  ang_o  = (bf16*)alloc((size_t)524288 * 64 * 2);
  bf16*  o1_act = (bf16*)alloc(512 * 256 * 2);
  bf16*  o2_act = (bf16*)alloc(512 * 128 * 2);
  float* xh_o   = (float*)alloc(512 * 16 * 4);
  float* meanb  = (float*)alloc(16 * 3 * 4);

  auto SM = [](const void* p, int kind, int kc, int ld, int md) {
    Seg s; s.p = p; s.kind = kind; s.kcols = kc; s.ld = ld; s.maskdiag = md;
    return s;
  };
  auto run = [&](std::initializer_list<Seg> segs, int rows, int Nout,
                 const bf16* Wt, const float* bias, int act, int maskp,
                 float* outf, bf16* outb) {
    GemmCfg c{};
    int K = 0, i = 0;
    for (const Seg& s : segs) { c.seg[i++] = s; K += s.kcols; }
    c.nseg = i; c.rows = rows; c.Nout = Nout; c.Wt = Wt; c.bias = bias;
    c.act = act; c.mask_pair = maskp; c.outf = outf; c.outb = outb;
    c.tvec = tvec;
    size_t wbytes = (size_t)K * Nout * sizeof(bf16);
    c.use_lds = (wbytes <= 65536) ? 1 : 0;
    size_t lds = c.use_lds ? wbytes : 0;
    const int NT = (Nout % 64 == 0) ? 4 : 1;
    int waves = (rows >> 4) * (Nout / (16 * NT));
    int blocks = (waves + 7) >> 3;  // 8 waves / block
#define LCH(KTV)                                                      \
  do {                                                                \
    if (NT == 4) k_wmma_gemm<KTV, 4><<<blocks, 256, lds, stream>>>(c);\
    else         k_wmma_gemm<KTV, 1><<<blocks, 256, lds, stream>>>(c);\
  } while (0)
    switch (K >> 5) {
      case 1:  LCH(1);  break;
      case 2:  LCH(2);  break;
      case 4:  LCH(4);  break;
      case 8:  LCH(8);  break;
      case 12: LCH(12); break;
      default: break;
    }
#undef LCH
  };

  // ---- initial embeddings & geometry ----
  k_embed<<<2, 256, 0, stream>>>(z_t, tvec, pos0, pos, Xemb);
  run({SM(Xemb, 0, 32, 32, 0)}, 512, 128, Wxh, xh_b, 1, 0, h_f, h_bm);
  k_radial<<<64, 256, 0, stream>>>(pos, radial, cd, vn);
  k_angle<<<2048, 256, 0, stream>>>(vn, ang);
  run({SM(radial, 3, 64, 0, 0)}, 16384, 128, Wde, edge_b, 0, 0, nullptr, dist_b);
  run({SM(h_bm, 2, 128, 128, 1), SM(h_bm, 1, 128, 128, 1),
       SM(dist_b, 0, 128, 128, 0)},
      16384, 128, Wnpf, npf_b, 0, 1, nullptr, pair_b);
  run({SM(ang, 4, 64, 0, 0)}, 524288, 64, Wae, angle_b, 1, 0, nullptr, tri_a);

  bf16* tcur = tri_a;
  bf16* tnxt = tri_b2;
  for (int l = 0; l < 4; ++l) {
    // edge MLP: concat(h_j, h_i, pair) -> HID -> swish -> EMB -> swish
    run({SM(h_bm, 2, 128, 128, 0), SM(h_bm, 1, 128, 128, 0),
         SM(pair_b, 0, 128, 128, 0)},
        16384, 256, We1[l], e1b[l], 2, 0, nullptr, e1_act);
    run({SM(e1_act, 0, 256, 256, 0)}, 16384, 128, We2[l], e2b[l], 2, 0, m_f,
        nullptr);
    // triangle aggregation + tri linear
    k_tsum<<<4096, 256, 0, stream>>>(tcur, ts_b);
    run({SM(ts_b, 0, 64, 64, 0)}, 16384, 128, Wtri[l], trb[l], 0, 0, tl_f,
        nullptr);
    k_pairnew<<<8192, 256, 0, stream>>>(m_f, tl_f, pair_b);
    // equivariant position update (uses cd from current geometry)
    k_coord<<<512, 32, 0, stream>>>(m_f, cd, co_w[l], co_b[l], pos);
    // node update
    k_agg<<<256, 256, 0, stream>>>(pair_b, agg_b);
    run({SM(h_bm, 0, 128, 128, 0), SM(agg_b, 0, 128, 128, 0)}, 512, 256,
        Wn1[l], n1b[l], 2, 0, nullptr, n1_act);
    run({SM(n1_act, 0, 256, 256, 0)}, 512, 128, Wn2[l], n2b[l], 0, 0, hd_f,
        nullptr);
    k_hupd<<<256, 256, 0, stream>>>(h_f, hd_f, h_bm);
    if (l < 3) {  // last layer's pair/tri recompute is dead code -> skipped
      k_radial<<<64, 256, 0, stream>>>(pos, radial, cd, vn);
      k_angle<<<2048, 256, 0, stream>>>(vn, ang);
      run({SM(ang, 4, 64, 0, 0)}, 524288, 64, Wae, angle_b, 0, 0, nullptr,
          ang_o);
      run({SM(ang_o, 0, 64, 64, 0), SM(tcur, 0, 64, 64, 0)}, 524288, 64, Wtef,
          tef_b, 1, 0, nullptr, tnxt);
      { bf16* t = tcur; tcur = tnxt; tnxt = t; }
      run({SM(radial, 3, 64, 0, 0)}, 16384, 128, Wde, edge_b, 0, 0, nullptr,
          dist_b);
      run({SM(dist_b, 0, 128, 128, 0), SM(pair_b, 0, 128, 128, 0)}, 16384, 128,
          Weef, eef_b, 1, 0, nullptr, pe1_b);
      run({SM(h_bm, 2, 128, 128, 1), SM(h_bm, 1, 128, 128, 1),
           SM(pe1_b, 0, 128, 128, 0)},
          16384, 128, Wnpf, npf_b, 0, 1, nullptr, pair_b);
    }
  }

  // ---- output head ----
  run({SM(h_bm, 0, 128, 128, 0)}, 512, 256, Wo1, o1b, 1, 0, nullptr, o1_act);
  run({SM(o1_act, 0, 256, 256, 0)}, 512, 128, Wo2, o2b, 1, 0, nullptr, o2_act);
  run({SM(o2_act, 0, 128, 128, 0)}, 512, 16, Wo3, nullptr, 0, 0, xh_o, nullptr);
  k_velmean<<<16, 32, 0, stream>>>(pos, pos0, meanb);
  k_final<<<2, 256, 0, stream>>>(pos, pos0, meanb, xh_o, (float*)d_out);
}
